// SelfAttention_43181601194826
// MI455X (gfx1250) — matrix operand order repaired and checked by reading, not compiled, not run on hardware
//
#include <hip/hip_runtime.h>
#include <hip/hip_bf16.h>

// ---------------------------------------------------------------------------
// Self-attention for MI455X (gfx1250), wave32 + WMMA bf16.
//   All GEMM B-operands are stored K-contiguous ([N,K] row-major):
//     - weights transposed once at f32->bf16 conversion
//     - K is naturally K-contiguous for Q@K^T
//     - V^T produced by a transposed-store epilogue of the V projection
//   LDS staging: gfx1250 async Global->LDS b128 copies (ASYNCcnt) when the
//   builtin is available, else a double-buffered reg pipeline. WMMA fragments
//   are ds_load_b128 pairs holding k = 16g..16g+15 per half-wave, same order for A and B.
// ---------------------------------------------------------------------------

typedef __attribute__((ext_vector_type(16))) __bf16 v16bf;
typedef __attribute__((ext_vector_type(8)))  float  v8f;

#define BM 64
#define BN 128
#define BK 32
#define THREADS 256

#if __has_builtin(__builtin_amdgcn_global_load_async_to_lds_b128)
#define USE_ASYNC_LDS 1
// Builtin signature (from hipcc diagnostic): first param is
//   __attribute__((vector_size(16))) int  __attribute__((address_space(1))) *
typedef __attribute__((__vector_size__(16))) int v4i_t;
typedef __attribute__((address_space(1))) v4i_t gv4i;
typedef __attribute__((address_space(3))) v4i_t lv4i;
#else
#define USE_ASYNC_LDS 0
#endif

__device__ __forceinline__ void wait_asynccnt0() {
#if __has_builtin(__builtin_amdgcn_s_wait_asynccnt)
  __builtin_amdgcn_s_wait_asynccnt(0);
#elif defined(__gfx1250__)
  asm volatile("s_wait_asynccnt 0x0" ::: "memory");
#endif
}

__device__ __forceinline__ unsigned short f32_to_bf16_rne(float f) {
  unsigned u = __float_as_uint(f);
  u += 0x7FFFu + ((u >> 16) & 1u);
  return (unsigned short)(u >> 16);
}

__global__ __launch_bounds__(256)
void cvt_f32_bf16_kernel(const float* __restrict__ in,
                         unsigned short* __restrict__ out, long n) {
  long i = (long)blockIdx.x * blockDim.x + threadIdx.x;
  long stride = (long)gridDim.x * blockDim.x;
  for (; i < n; i += stride) out[i] = f32_to_bf16_rne(in[i]);
}

// in: [R,C] f32 row-major  ->  out: [C,R] bf16 row-major (tiled transpose)
__global__ __launch_bounds__(256)
void cvt_transpose_bf16_kernel(const float* __restrict__ in,
                               unsigned short* __restrict__ out,
                               int R, int C) {
  __shared__ float tile[32][33];
  const int bx = blockIdx.x * 32;   // column base (C dim)
  const int by = blockIdx.y * 32;   // row base (R dim)
  const int tx = threadIdx.x & 31;
  const int ty = threadIdx.x >> 5;  // 0..7
#pragma unroll
  for (int i = 0; i < 32; i += 8)
    tile[ty + i][tx] = in[(long)(by + ty + i) * C + (bx + tx)];
  __syncthreads();
#pragma unroll
  for (int i = 0; i < 32; i += 8)
    out[(long)(bx + ty + i) * R + (by + tx)] = f32_to_bf16_rne(tile[tx][ty + i]);
}

// ---------------------------------------------------------------------------
// Tiled bf16 GEMM: C = scale * (A @ B) + bias.
//  A: [M,K] bf16 row-major, lda
//  B: stored [N,K] bf16 row-major (K-contiguous), ldb
//  C: OUTF32 -> float [M,N]; else bf16; OUTT -> bf16 stored transposed [N,M]
//  CAUSAL: epilogue writes -1e9 where col > row (per-batch coordinates)
//  causal_trunc: limits the K loop to keys <= max row of the tile
// ---------------------------------------------------------------------------
template <bool OUTF32, bool CAUSAL, bool OUTT>
__global__ __launch_bounds__(THREADS)
void gemm_bf16_wmma(const unsigned short* __restrict__ A,
                    const unsigned short* __restrict__ Bm,
                    void* __restrict__ Cv,
                    const float* __restrict__ bias,
                    float scale,
                    int M, int N, int K,
                    int lda, int ldb, int ldc,
                    long sA, long sB, long sC,
                    int causal_trunc) {
  __shared__ __align__(16) unsigned short lsA[2][BM * BK];
  __shared__ __align__(16) unsigned short lsB[2][BN * BK];

  const int bz = blockIdx.z;
  const unsigned short* Ab = A  + (long)bz * sA;
  const unsigned short* Bb = Bm + (long)bz * sB;
  const int m0 = blockIdx.y * BM;
  const int n0 = blockIdx.x * BN;
  const int tid  = threadIdx.x;
  const int wave = tid >> 5;
  const int lane = tid & 31;
  const int g    = lane >> 4;           // half-wave group
  const int ln   = lane & 15;
  const int colb = n0 + wave * 16 + ln; // this lane's output column

  if (CAUSAL && n0 > m0 + BM - 1) {     // tile entirely above the diagonal
    float* Cb = (float*)Cv + (long)bz * sC;
#pragma unroll
    for (int s = 0; s < 4; ++s)
#pragma unroll
      for (int r = 0; r < 8; ++r) {
        int row = m0 + s * 16 + 8 * g + r;
        Cb[(long)row * ldc + colb] = -1e9f;
      }
    return;
  }

  const v8f vzero = {0.f, 0.f, 0.f, 0.f, 0.f, 0.f, 0.f, 0.f};
  v8f acc[4] = {vzero, vzero, vzero, vzero};

  int kEnd = K;
  if (causal_trunc) {
    kEnd = m0 + BM;
    kEnd = (kEnd + BK - 1) & ~(BK - 1);
    if (kEnd > K) kEnd = K;
  }
  const int nsteps = kEnd / BK;

  // staging coordinates (per thread): A one b128, B two b128 (same n-row)
  const int arow = tid >> 2;
  const int acol = (tid & 3) * 8;
  const int bnr  = tid >> 1;
  const int bkc  = (tid & 1) * 16;

#if USE_ASYNC_LDS
  // -------- async Global->LDS staging (ASYNCcnt) --------
  auto asyncStage = [&](int k0, int buf) {
    __builtin_amdgcn_global_load_async_to_lds_b128(
        (gv4i*)(Ab + (long)(m0 + arow) * lda + (k0 + acol)),
        (lv4i*)&lsA[buf][arow * BK + acol], 0, 0);
    __builtin_amdgcn_global_load_async_to_lds_b128(
        (gv4i*)(Bb + (long)(n0 + bnr) * ldb + (k0 + bkc)),
        (lv4i*)&lsB[buf][bnr * BK + bkc], 0, 0);
    __builtin_amdgcn_global_load_async_to_lds_b128(
        (gv4i*)(Bb + (long)(n0 + bnr) * ldb + (k0 + bkc + 8)),
        (lv4i*)&lsB[buf][bnr * BK + bkc + 8], 0, 0);
  };
  asyncStage(0, 0);
  wait_asynccnt0();
  __syncthreads();
#else
  // -------- register-staged double-buffer pipeline --------
  uint4 ra, rb0, rb1;
  auto loadG = [&](int k0) {
    ra  = *(const uint4*)(Ab + (long)(m0 + arow) * lda + (k0 + acol));
    rb0 = *(const uint4*)(Bb + (long)(n0 + bnr)  * ldb + (k0 + bkc));
    rb1 = *(const uint4*)(Bb + (long)(n0 + bnr)  * ldb + (k0 + bkc + 8));
  };
  auto storeL = [&](int buf) {
    *(uint4*)&lsA[buf][arow * BK + acol]    = ra;
    *(uint4*)&lsB[buf][bnr  * BK + bkc]     = rb0;
    *(uint4*)&lsB[buf][bnr  * BK + bkc + 8] = rb1;
  };
  loadG(0);
  storeL(0);
  __syncthreads();
#endif

  for (int s = 0; s < nsteps; ++s) {
    const int cur = s & 1;
    if (s + 1 < nsteps) {
#if USE_ASYNC_LDS
      asyncStage((s + 1) * BK, cur ^ 1);
#else
      loadG((s + 1) * BK);
#endif
      if (s + 2 < nsteps) {
        __builtin_prefetch(Ab + (long)(m0 + arow) * lda + ((s + 2) * BK + acol), 0, 0);
        __builtin_prefetch(Bb + (long)(n0 + bnr)  * ldb + ((s + 2) * BK + bkc), 0, 0);
      }
    }

    // B fragment (reused across the wave's 4 M-subtiles):
    //   lane ln = n; VGPR 0-7 hold k = 16g .. 16g+15  -> two b128
    union { uint4 u[2]; v16bf v; } fb;
    const unsigned short* bp = &lsB[cur][(wave * 16 + ln) * BK + 16 * g];
    fb.u[0] = *(const uint4*)bp;
    fb.u[1] = *(const uint4*)(bp + 8);

#pragma unroll
    for (int t = 0; t < 4; ++t) {
      // A fragment: lane ln = row; VGPR 0-7 hold k = 16g .. 16g+15
      union { uint4 u[2]; v16bf v; } fa;
      const unsigned short* ap = &lsA[cur][(t * 16 + ln) * BK + 16 * g];
      fa.u[0] = *(const uint4*)ap;
      fa.u[1] = *(const uint4*)(ap + 8);
      acc[t] = __builtin_amdgcn_wmma_f32_16x16x32_bf16(
          false, fa.v, false, fb.v, (short)0, acc[t], false, false);
    }

    if (s + 1 < nsteps) {
#if USE_ASYNC_LDS
      wait_asynccnt0();
#else
      storeL(cur ^ 1);
#endif
    }
    __syncthreads();
  }

  const float bcol = bias ? bias[colb] : 0.0f;
  if (OUTF32) {
    float* Cb = (float*)Cv + (long)bz * sC;
#pragma unroll
    for (int s = 0; s < 4; ++s)
#pragma unroll
      for (int r = 0; r < 8; ++r) {
        int row = m0 + s * 16 + 8 * g + r;
        float v = acc[s][r] * scale + bcol;
        if (CAUSAL && colb > row) v = -1e9f;
        Cb[(long)row * ldc + colb] = v;
      }
  } else if (OUTT) {
    unsigned short* Cb = (unsigned short*)Cv + (long)bz * sC;
#pragma unroll
    for (int s = 0; s < 4; ++s)
#pragma unroll
      for (int r = 0; r < 8; ++r) {
        int row = m0 + s * 16 + 8 * g + r;
        float v = acc[s][r] * scale + bcol;
        Cb[(long)colb * ldc + row] = f32_to_bf16_rne(v);   // transposed store
      }
  } else {
    unsigned short* Cb = (unsigned short*)Cv + (long)bz * sC;
#pragma unroll
    for (int s = 0; s < 4; ++s)
#pragma unroll
      for (int r = 0; r < 8; ++r) {
        int row = m0 + s * 16 + 8 * g + r;
        float v = acc[s][r] * scale + bcol;
        Cb[(long)row * ldc + colb] = f32_to_bf16_rne(v);
      }
  }
}

// ---------------------------------------------------------------------------
// Row softmax over 2048 f32, writing bf16 in place (bf16 row occupies the
// first half of the f32 row's bytes; one block owns a row -> no races).
// ---------------------------------------------------------------------------
__global__ __launch_bounds__(256)
void softmax_inplace_bf16(float* __restrict__ S) {
  __shared__ float red[256];
  const long row = blockIdx.x;
  float* sp = S + row * 2048l;
  unsigned short* pp = (unsigned short*)sp;
  const int t = threadIdx.x;

  float v[8];
  float mx = -3.4e38f;
#pragma unroll
  for (int i = 0; i < 8; ++i) {
    v[i] = sp[t + i * 256];
    mx = fmaxf(mx, v[i]);
  }
  red[t] = mx; __syncthreads();
  for (int s = 128; s > 0; s >>= 1) {
    if (t < s) red[t] = fmaxf(red[t], red[t + s]);
    __syncthreads();
  }
  mx = red[0]; __syncthreads();

  float sum = 0.f;
#pragma unroll
  for (int i = 0; i < 8; ++i) { v[i] = __expf(v[i] - mx); sum += v[i]; }
  red[t] = sum; __syncthreads();
  for (int s = 128; s > 0; s >>= 1) {
    if (t < s) red[t] += red[t + s];
    __syncthreads();
  }
  const float inv = 1.0f / red[0];
  __syncthreads();
#pragma unroll
  for (int i = 0; i < 8; ++i) pp[t + i * 256] = f32_to_bf16_rne(v[i] * inv);
}

// ---------------------------------------------------------------------------
extern "C" void kernel_launch(void* const* d_in, const int* in_sizes, int n_in,
                              void* d_out, int out_size, void* d_ws, size_t ws_size,
                              hipStream_t stream) {
  const int Bn = 4, S = 2048, H = 1024, Dd = 1024;
  const long MS = (long)Bn * S;          // 8192 rows total

  const float* x  = (const float*)d_in[0];
  // d_in[1] = causal_mask (handled analytically)
  const float* Wq = (const float*)d_in[2];
  const float* bq = (const float*)d_in[3];
  const float* Wk = (const float*)d_in[4];
  const float* bk = (const float*)d_in[5];
  const float* Wv = (const float*)d_in[6];
  const float* bv = (const float*)d_in[7];
  const float* Wo = (const float*)d_in[8];
  const float* bo = (const float*)d_in[9];

  char* ws = (char*)d_ws;
  size_t off = 0;
  auto alloc = [&](size_t bytes) -> void* {
    void* p = ws + off;
    off += (bytes + 255) & ~(size_t)255;
    return p;
  };
  unsigned short* xb   = (unsigned short*)alloc(MS * H * 2);          // 16 MB
  unsigned short* wqT  = (unsigned short*)alloc((size_t)H * Dd * 2);  // [D,H]
  unsigned short* wkT  = (unsigned short*)alloc((size_t)H * Dd * 2);  // [D,H]
  unsigned short* wvT  = (unsigned short*)alloc((size_t)H * Dd * 2);  // [D,H]
  unsigned short* woT  = (unsigned short*)alloc((size_t)Dd * H * 2);  // [H,D]
  unsigned short* Qb   = (unsigned short*)alloc(MS * Dd * 2);         // 16 MB
  unsigned short* Kb   = (unsigned short*)alloc(MS * Dd * 2);         // 16 MB
  unsigned short* Vt   = (unsigned short*)alloc(MS * Dd * 2);         // 16 MB, [b][D][S]
  float*          scrs = (float*)alloc((size_t)Bn * S * S * 4);       // 64 MB (P in place)
  unsigned short* attn = (unsigned short*)alloc(MS * Dd * 2);         // 16 MB

  // 1) conversions: x plain; weights transposed to K-contiguous layout
  cvt_f32_bf16_kernel<<<4096, 256, 0, stream>>>(x, xb, MS * (long)H);
  dim3 gT(Dd / 32, H / 32, 1);
  cvt_transpose_bf16_kernel<<<gT, 256, 0, stream>>>(Wq, wqT, H, Dd);
  cvt_transpose_bf16_kernel<<<gT, 256, 0, stream>>>(Wk, wkT, H, Dd);
  cvt_transpose_bf16_kernel<<<gT, 256, 0, stream>>>(Wv, wvT, H, Dd);
  dim3 gTo(H / 32, Dd / 32, 1);
  cvt_transpose_bf16_kernel<<<gTo, 256, 0, stream>>>(Wo, woT, Dd, H);

  // 2) Q/K projections: [8192,1024] x [1024,1024]^T-layout + bias -> bf16
  dim3 gQK(Dd / BN, MS / BM, 1);
  gemm_bf16_wmma<false, false, false><<<gQK, THREADS, 0, stream>>>(
      xb, wqT, Qb, bq, 1.0f, (int)MS, Dd, H, H, H, Dd, 0, 0, 0, 0);
  gemm_bf16_wmma<false, false, false><<<gQK, THREADS, 0, stream>>>(
      xb, wkT, Kb, bk, 1.0f, (int)MS, Dd, H, H, H, Dd, 0, 0, 0, 0);

  // V projection with transposed store -> Vt[b] = [D,S] (K-contiguous for P@V)
  dim3 gV(Dd / BN, S / BM, Bn);
  gemm_bf16_wmma<false, false, true><<<gV, THREADS, 0, stream>>>(
      xb, wvT, Vt, bv, 1.0f, S, Dd, H, H, H, S,
      (long)S * H, 0, (long)Dd * S, 0);

  // 3) scores = (Q @ K^T) * D^-1/2, causal mask folded into epilogue (f32)
  dim3 gS(S / BN, S / BM, Bn);
  gemm_bf16_wmma<true, true, false><<<gS, THREADS, 0, stream>>>(
      Qb, Kb, scrs, nullptr, 1.0f / 32.0f, S, S, Dd,
      Dd, Dd, S, (long)S * Dd, (long)S * Dd, (long)S * S, 0);

  // 4) row softmax, bf16 P written in place (P lda = 4096 ushorts)
  softmax_inplace_bf16<<<(unsigned)(Bn * S), 256, 0, stream>>>(scrs);

  // 5) attn @ V via Vt (K-contiguous), causal-truncated K loop -> bf16
  dim3 gPV(Dd / BN, S / BM, Bn);
  gemm_bf16_wmma<false, false, false><<<gPV, THREADS, 0, stream>>>(
      (const unsigned short*)scrs, Vt, attn, nullptr, 1.0f, S, Dd, S,
      2 * S /*lda=4096*/, S, Dd,
      (long)S * 2 * S, (long)Dd * S, (long)S * Dd, 1);

  // 6) final projection: [8192,1024] x Wo^T-layout + bo -> f32 d_out
  dim3 gO(H / BN, MS / BM, 1);
  gemm_bf16_wmma<true, false, false><<<gO, THREADS, 0, stream>>>(
      attn, woT, d_out, bo, 1.0f, (int)MS, H, Dd, Dd, Dd, H, 0, 0, 0, 0);
}
